// Block_56427280335230
// MI455X (gfx1250) — compile-verified, matrix operand order checked by reading, not run on hardware
//
#include <hip/hip_runtime.h>
#include <hip/hip_bf16.h>
#include <math.h>

typedef __attribute__((ext_vector_type(16))) _Float16 v16h;
typedef __attribute__((ext_vector_type(8)))  _Float16 v8h;
typedef __attribute__((ext_vector_type(8)))  float    v8f;
typedef unsigned int u32x4 __attribute__((ext_vector_type(4)));
typedef int          i32x8 __attribute__((ext_vector_type(8)));
typedef int          i32x4 __attribute__((ext_vector_type(4)));

#define B_    2
#define L_    2048
#define H_    16
#define D_    64
#define HID_  1024
#define QKVP_ 7168
#define VP_   5120
#define ACC_  128
#define MTOT  (B_ * L_)   // 4096

// ---------------------------------------------------------------------------
// helpers
// ---------------------------------------------------------------------------

// 16-half WMMA A/B fragment for lane-half `half` (pointer pre-offset by
// half*8 halves): elements 0..7 <-> K=half*8+0..7, 8..15 <-> K=16+half*8+0..7.
__device__ __forceinline__ v16h ldfrag(const _Float16* p) {
    v8h lo = *reinterpret_cast<const v8h*>(p);
    v8h hi = *reinterpret_cast<const v8h*>(p + 16);
    v16h r;
#pragma unroll
    for (int i = 0; i < 8; ++i) { r[i] = lo[i]; r[8 + i] = hi[i]; }
    return r;
}

__device__ __forceinline__ v8f wmma32(v16h a, v16h b, v8f c) {
    return __builtin_amdgcn_wmma_f32_16x16x32_f16(
        false, a, false, b, (short)0, c, false, false);
}

__device__ __forceinline__ float rmax16(float v) {
#pragma unroll
    for (int m = 1; m < 16; m <<= 1) v = fmaxf(v, __shfl_xor(v, m, 16));
    return v;
}
__device__ __forceinline__ float rsum16(float v) {
#pragma unroll
    for (int m = 1; m < 16; m <<= 1) v += __shfl_xor(v, m, 16);
    return v;
}

// Branchless exact-GELU (erf via Abramowitz-Stegun 7.1.26, |err|~1.5e-7).
__device__ __forceinline__ float gelu_exact(float v) {
    float x  = v * 0.70710678118f;
    float ax = fabsf(x);
    float t  = 1.0f / (1.0f + 0.3275911f * ax);
    float poly = ((((1.061405429f * t - 1.453152027f) * t + 1.421413741f) * t
                   - 0.284496736f) * t + 0.254829592f) * t;
    float y  = 1.0f - poly * __expf(-ax * ax);
    return 0.5f * v * (1.0f + copysignf(y, x));
}

// ---------------------------------------------------------------------------
// TDM: issue a 32-row x 32-col f16 tile load (line stride Kelems) into LDS.
// D# built per CDNA5 ISA 8.3/8.4: group0 = {count/type, lds_addr, 57b
// global_addr}, group1 = {data_size=2B, tensor_dim0=K, tensor_dim1=32,
// tile_dim0=32, tile_dim1=32, tensor_dim0_stride=K}. Groups 2/3 zero (2D).
// ---------------------------------------------------------------------------
__device__ __forceinline__ void tdm_load_tile(const _Float16* gptr,
                                              unsigned lds_addr, int Kelems) {
    unsigned long long ga = (unsigned long long)(uintptr_t)gptr;
    u32x4 g0;
    g0.x = (unsigned)__builtin_amdgcn_readfirstlane(1u);
    g0.y = (unsigned)__builtin_amdgcn_readfirstlane(lds_addr);
    g0.z = (unsigned)__builtin_amdgcn_readfirstlane((unsigned)ga);
    g0.w = (unsigned)__builtin_amdgcn_readfirstlane(
        (unsigned)((ga >> 32) & 0x01FFFFFFull) | (2u << 30));
    i32x8 g1;
    g1[0] = (int)(1u << 16);                         // data_size = 2 bytes
    g1[1] = (int)((Kelems & 0xFFFF) << 16);          // tensor_dim0 lo
    g1[2] = (int)((Kelems >> 16) | (32 << 16));      // tensor_dim0 hi | dim1 lo
    g1[3] = (int)(32 << 16);                         // tile_dim0 = 32
    g1[4] = 32;                                      // tile_dim1 = 32
    g1[5] = Kelems;                                  // tensor_dim0_stride lo
    g1[6] = 0;
    g1[7] = 0;
    i32x4 gz = {0, 0, 0, 0};
#if __clang_major__ >= 23
    i32x8 gz8 = {0, 0, 0, 0, 0, 0, 0, 0};
    __builtin_amdgcn_tensor_load_to_lds(g0, g1, gz, gz, gz8, 0);
#else
    __builtin_amdgcn_tensor_load_to_lds(g0, g1, gz, gz, 0);
#endif
}

// ---------------------------------------------------------------------------
// 0) f32 -> f16 convert
// ---------------------------------------------------------------------------
__global__ void cvt_f16_kernel(const float* __restrict__ in,
                               _Float16* __restrict__ out, int n) {
    int i = blockIdx.x * blockDim.x + threadIdx.x;
    if (i < n) out[i] = (_Float16)in[i];
}

// ---------------------------------------------------------------------------
// 1) LayerNorm: one row (1024 f32) per 256-thread block
// ---------------------------------------------------------------------------
__global__ __launch_bounds__(256) void ln_kernel(
    const float* __restrict__ x, const float* __restrict__ lw,
    const float* __restrict__ lb, _Float16* __restrict__ h16,
    float* __restrict__ hacc) {
    int row = blockIdx.x;
    int tid = threadIdx.x;
    const float4 v = reinterpret_cast<const float4*>(x + (size_t)row * HID_)[tid];
    float s = v.x + v.y + v.z + v.w;
    float q = v.x * v.x + v.y * v.y + v.z * v.z + v.w * v.w;
#pragma unroll
    for (int m = 16; m >= 1; m >>= 1) { s += __shfl_xor(s, m); q += __shfl_xor(q, m); }
    __shared__ float rs[8], rq[8];
    if ((tid & 31) == 0) { rs[tid >> 5] = s; rq[tid >> 5] = q; }
    __syncthreads();
    float ts = 0.f, tq = 0.f;
#pragma unroll
    for (int i = 0; i < 8; ++i) { ts += rs[i]; tq += rq[i]; }
    const float mu = ts * (1.f / HID_);
    const float var = tq * (1.f / HID_) - mu * mu;
    const float rstd = rsqrtf(var + 1e-5f);
    const float xs[4] = {v.x, v.y, v.z, v.w};
    int c0 = tid * 4;
#pragma unroll
    for (int j = 0; j < 4; ++j) {
        int c = c0 + j;
        float y = (xs[j] - mu) * rstd * lw[c] + lb[c];
        h16[(size_t)row * HID_ + c] = (_Float16)y;
        if (c < ACC_) hacc[(size_t)row * ACC_ + c] = y;
    }
}

// ---------------------------------------------------------------------------
// 2) cumlogsumexp scan over L (stable recurrence), one thread per (b,c)
// ---------------------------------------------------------------------------
__global__ __launch_bounds__(256) void scan_kernel(
    const float* __restrict__ hacc, _Float16* __restrict__ h16) {
    int tid = blockIdx.x * blockDim.x + threadIdx.x;
    if (tid >= B_ * ACC_) return;
    int b = tid >> 7;
    int c = tid & (ACC_ - 1);
    float M = -1e30f, S = 0.f;
    for (int t = 0; t < L_; ++t) {
        size_t row = (size_t)(b * L_ + t);
        float a = 5.f * hacc[row * ACC_ + c];
        float nM = fmaxf(M, a);
        S = S * __expf(M - nM) + __expf(a - nM);
        M = nM;
        h16[row * HID_ + c] = (_Float16)((M + logf(S)) * 0.2f);
    }
}

// ---------------------------------------------------------------------------
// Shared GEMM mainloop body: one wave per block, 64x32 tile (4x2 WMMAs),
// A fragments from global (+prefetch), B 32x32 tile TDM-staged into an LDS
// double buffer, fragments via ds_load_b128. Accumulators stay in-place.
// ---------------------------------------------------------------------------
#define GEMM_MAINLOOP(A_, W_, KDIM)                                           \
    __shared__ __align__(16) _Float16 bstage[2][32 * 32];                     \
    const int K = (KDIM);                                                     \
    int lane = threadIdx.x & 31;                                              \
    int half = lane >> 4, lr = lane & 15;                                     \
    int wid = blockIdx.x;                                                     \
    const int MT = MTOT / 64;                                                 \
    int m0 = (wid % MT) * 64;                                                 \
    int n0 = (wid / MT) * 32;                                                 \
    v8f acc[4][2];                                                            \
    _Pragma("unroll") for (int mi = 0; mi < 4; ++mi)                          \
        _Pragma("unroll") for (int ni = 0; ni < 2; ++ni)                      \
            acc[mi][ni] = (v8f){0.f, 0.f, 0.f, 0.f, 0.f, 0.f, 0.f, 0.f};      \
    const _Float16* arow[4];                                                  \
    _Pragma("unroll") for (int mi = 0; mi < 4; ++mi)                          \
        arow[mi] = (A_) + (size_t)(m0 + mi * 16 + lr) * K + half * 8;         \
    const _Float16* wtile = (W_) + (size_t)n0 * K;                            \
    unsigned lds0 = (unsigned)(uintptr_t)(&bstage[0][0]);                     \
    tdm_load_tile(wtile, lds0, K);                                            \
    int buf = 0;                                                              \
    for (int k = 0; k < K; k += 32) {                                         \
        v16h af[4];                                                           \
        _Pragma("unroll") for (int mi = 0; mi < 4; ++mi) {                    \
            af[mi] = ldfrag(arow[mi] + k);                                    \
            __builtin_prefetch(arow[mi] + k + 128, 0, 3);                     \
        }                                                                     \
        if (k + 32 < K) {                                                     \
            asm volatile("s_wait_dscnt 0x0" ::: "memory");                    \
            tdm_load_tile(wtile + k + 32, lds0 + (buf ^ 1) * 2048, K);        \
            __builtin_amdgcn_s_wait_tensorcnt((short)1);                      \
        } else {                                                              \
            __builtin_amdgcn_s_wait_tensorcnt((short)0);                      \
        }                                                                     \
        asm volatile("" ::: "memory");                                        \
        v16h bf[2];                                                           \
        _Pragma("unroll") for (int ni = 0; ni < 2; ++ni)                      \
            bf[ni] = ldfrag(&bstage[buf][(ni * 16 + lr) * 32 + half * 8]);    \
        _Pragma("unroll") for (int mi = 0; mi < 4; ++mi)                      \
            _Pragma("unroll") for (int ni = 0; ni < 2; ++ni)                  \
                acc[mi][ni] = wmma32(af[mi], bf[ni], acc[mi][ni]);            \
        buf ^= 1;                                                             \
    }

// ---------------------------------------------------------------------------
// 3) GEMM1: qkvp = h @ w_in^T  (M=4096, N=7168, K=1024). Fused epilogue:
//    q,k -> f32 (RoPE), v -> f16 transposed vT[b][h][d][l], p -> GELU f16.
// ---------------------------------------------------------------------------
__global__ __launch_bounds__(32) void gemm_qkvp_kernel(
    const _Float16* __restrict__ A, const _Float16* __restrict__ W,
    float* __restrict__ qf32, float* __restrict__ kf32,
    _Float16* __restrict__ vT, _Float16* __restrict__ g2in) {
    GEMM_MAINLOOP(A, W, HID_)
#pragma unroll
    for (int mi = 0; mi < 4; ++mi)
#pragma unroll
        for (int ni = 0; ni < 2; ++ni)
#pragma unroll
            for (int r = 0; r < 8; ++r) {
                int row = m0 + mi * 16 + r + half * 8;
                int col = n0 + ni * 16 + lr;
                float v = acc[mi][ni][r];
                if (col < HID_) {
                    qf32[(size_t)row * HID_ + col] = v;
                } else if (col < 2 * HID_) {
                    kf32[(size_t)row * HID_ + (col - HID_)] = v;
                } else if (col < 3 * HID_) {
                    int c = col - 2 * HID_;          // h*64 + d
                    int bb = row >> 11, ll = row & (L_ - 1);
                    vT[((size_t)(bb * H_ + (c >> 6)) * D_ + (c & 63)) * L_ + ll] =
                        (_Float16)v;
                } else {
                    g2in[(size_t)row * VP_ + HID_ + (col - 3 * HID_)] =
                        (_Float16)gelu_exact(v);
                }
            }
}

// ---------------------------------------------------------------------------
// 4) RoPE on q,k: one thread per (b,l,h,i<32), rotate pair (i, i+32)
// ---------------------------------------------------------------------------
__global__ __launch_bounds__(256) void rope_kernel(
    const float* __restrict__ qf32, const float* __restrict__ kf32,
    _Float16* __restrict__ q16, _Float16* __restrict__ k16) {
    int idx = blockIdx.x * blockDim.x + threadIdx.x;  // < 2^21
    int i = idx & 31;
    int hh = (idx >> 5) & 15;
    int l = (idx >> 9) & (L_ - 1);
    int b = idx >> 20;
    float ang = (float)l * __expf(-0.2878231366f * (float)i); // 10000^(-i/32)
    float sn, cs;
    __sincosf(ang, &sn, &cs);
    size_t base = (size_t)(b * L_ + l) * HID_ + hh * 64 + i;
    float q0 = qf32[base], q1 = qf32[base + 32];
    q16[base]      = (_Float16)(q0 * cs - q1 * sn);
    q16[base + 32] = (_Float16)(q1 * cs + q0 * sn);
    float k0 = kf32[base], k1 = kf32[base + 32];
    k16[base]      = (_Float16)(k0 * cs - k1 * sn);
    k16[base + 32] = (_Float16)(k1 * cs + k0 * sn);
}

// ---------------------------------------------------------------------------
// 5) Causal flash attention: one wave per 16-row q tile; S via 2 chained
//    WMMAs, online softmax in 16-lane halves, P relayout via LDS, PV via 4
//    WMMAs per 32 keys reading pre-transposed vT contiguously.
// ---------------------------------------------------------------------------
__global__ __launch_bounds__(32) void attn_kernel(
    const _Float16* __restrict__ q16, const _Float16* __restrict__ k16,
    const _Float16* __restrict__ vT, _Float16* __restrict__ g2in) {
    __shared__ __align__(16) _Float16 pbuf[16 * 32];

    int lane = threadIdx.x & 31;
    int half = lane >> 4, lr = lane & 15;
    int wid = blockIdx.x;
    int qt = wid & 127;
    int hh = (wid >> 7) & 15;
    int b  = wid >> 11;
    int qbase = qt * 16;

    size_t rowQ = (size_t)(b * L_ + qbase + lr) * HID_ + hh * 64;
    v16h qa0 = ldfrag(q16 + rowQ + half * 8);
    v16h qa1 = ldfrag(q16 + rowQ + 32 + half * 8);

    const _Float16* vbase = vT + (size_t)(b * H_ + hh) * D_ * L_;

    float m[8], lsum[8];
    v8f  acc[4];
#pragma unroll
    for (int r = 0; r < 8; ++r) { m[r] = -1e30f; lsum[r] = 0.f; }
#pragma unroll
    for (int t = 0; t < 4; ++t)
        acc[t] = (v8f){0.f, 0.f, 0.f, 0.f, 0.f, 0.f, 0.f, 0.f};

    const int npair = (qt + 2) >> 1;
    for (int p = 0; p < npair; ++p) {
        int j0 = p * 32;
        v8f s0 = (v8f){0.f, 0.f, 0.f, 0.f, 0.f, 0.f, 0.f, 0.f};
        v8f s1 = s0;

        size_t rowK0 = (size_t)(b * L_ + j0 + lr) * HID_ + hh * 64;
        __builtin_prefetch(k16 + rowK0 + (size_t)32 * HID_, 0, 3);
        v16h kb0 = ldfrag(k16 + rowK0 + half * 8);
        v16h kb1 = ldfrag(k16 + rowK0 + 32 + half * 8);
        s0 = wmma32(qa0, kb0, s0);
        s0 = wmma32(qa1, kb1, s0);
        size_t rowK1 = rowK0 + (size_t)16 * HID_;
        kb0 = ldfrag(k16 + rowK1 + half * 8);
        kb1 = ldfrag(k16 + rowK1 + 32 + half * 8);
        s1 = wmma32(qa0, kb0, s1);
        s1 = wmma32(qa1, kb1, s1);

#pragma unroll
        for (int r = 0; r < 8; ++r) {
            int qrow = qbase + r + half * 8;
            float e0 = s0[r] * 0.125f;
            float e1 = s1[r] * 0.125f;
            if (j0 + lr > qrow)      e0 = -1e30f;
            if (j0 + 16 + lr > qrow) e1 = -1e30f;
            float nm = fmaxf(m[r], fmaxf(rmax16(e0), rmax16(e1)));
            float alpha = __expf(m[r] - nm);
            float p0 = __expf(e0 - nm);
            float p1 = __expf(e1 - nm);
            lsum[r] = lsum[r] * alpha + rsum16(p0) + rsum16(p1);
            m[r] = nm;
#pragma unroll
            for (int t = 0; t < 4; ++t) acc[t][r] *= alpha;
            int prow = r + half * 8;
            pbuf[prow * 32 + lr]      = (_Float16)p0;
            pbuf[prow * 32 + 16 + lr] = (_Float16)p1;
        }
        __syncthreads();  // single-wave WG: LDS ordering fence only

        v8h plo = *reinterpret_cast<const v8h*>(&pbuf[lr * 32 + half * 8]);
        v8h phi = *reinterpret_cast<const v8h*>(&pbuf[lr * 32 + 16 + half * 8]);
        v16h pA;
#pragma unroll
        for (int i = 0; i < 8; ++i) { pA[i] = plo[i]; pA[8 + i] = phi[i]; }
        __syncthreads();

#pragma unroll
        for (int t = 0; t < 4; ++t) {
            const _Float16* vp = vbase + (size_t)(t * 16 + lr) * L_ + j0;
            __builtin_prefetch(vp + 32, 0, 3);
            v16h vb = ldfrag(vp + half * 8);
            acc[t] = wmma32(pA, vb, acc[t]);
        }
    }

    float inv[8];
#pragma unroll
    for (int r = 0; r < 8; ++r) inv[r] = 1.0f / lsum[r];
#pragma unroll
    for (int t = 0; t < 4; ++t)
#pragma unroll
        for (int r = 0; r < 8; ++r) {
            int row = qbase + r + half * 8;
            int col = hh * 64 + t * 16 + lr;
            g2in[(size_t)(b * L_ + row) * VP_ + col] =
                (_Float16)(acc[t][r] * inv[r]);
        }
}

// ---------------------------------------------------------------------------
// 6) GEMM2: out = g2in @ w_out^T + b_out  (M=4096, N=1024, K=5120)
// ---------------------------------------------------------------------------
__global__ __launch_bounds__(32) void gemm_out_kernel(
    const _Float16* __restrict__ A, const _Float16* __restrict__ W,
    const float* __restrict__ bout, float* __restrict__ out) {
    GEMM_MAINLOOP(A, W, VP_)
#pragma unroll
    for (int mi = 0; mi < 4; ++mi)
#pragma unroll
        for (int ni = 0; ni < 2; ++ni)
#pragma unroll
            for (int r = 0; r < 8; ++r) {
                int row = m0 + mi * 16 + r + half * 8;
                int col = n0 + ni * 16 + lr;
                out[(size_t)row * HID_ + col] = acc[mi][ni][r] + bout[col];
            }
}

// ---------------------------------------------------------------------------
// host launcher
// ---------------------------------------------------------------------------
extern "C" void kernel_launch(void* const* d_in, const int* in_sizes, int n_in,
                              void* d_out, int out_size, void* d_ws,
                              size_t ws_size, hipStream_t stream) {
    const float* x     = (const float*)d_in[0];
    const float* ln_w  = (const float*)d_in[1];
    const float* ln_b  = (const float*)d_in[2];
    const float* w_in  = (const float*)d_in[3];
    const float* w_out = (const float*)d_in[4];
    const float* b_out = (const float*)d_in[5];
    float* out = (float*)d_out;

    char* ws = (char*)d_ws;
    size_t off = 0;
    auto carve = [&](size_t bytes) {
        void* p = ws + off;
        off += (bytes + 255) & ~(size_t)255;
        return p;
    };
    float*    hacc = (float*)carve((size_t)MTOT * ACC_ * 4);
    _Float16* h16  = (_Float16*)carve((size_t)MTOT * HID_ * 2);
    _Float16* wi16 = (_Float16*)carve((size_t)QKVP_ * HID_ * 2);
    _Float16* wo16 = (_Float16*)carve((size_t)HID_ * VP_ * 2);
    float*    qf32 = (float*)carve((size_t)MTOT * HID_ * 4);
    float*    kf32 = (float*)carve((size_t)MTOT * HID_ * 4);
    _Float16* q16  = (_Float16*)carve((size_t)MTOT * HID_ * 2);
    _Float16* k16  = (_Float16*)carve((size_t)MTOT * HID_ * 2);
    _Float16* vT   = (_Float16*)carve((size_t)B_ * H_ * D_ * L_ * 2);
    _Float16* g2in = (_Float16*)carve((size_t)MTOT * VP_ * 2);
    (void)ws_size; (void)in_sizes; (void)n_in; (void)out_size;

    {
        int n1 = QKVP_ * HID_;
        cvt_f16_kernel<<<(n1 + 255) / 256, 256, 0, stream>>>(w_in, wi16, n1);
        int n2 = HID_ * VP_;
        cvt_f16_kernel<<<(n2 + 255) / 256, 256, 0, stream>>>(w_out, wo16, n2);
    }
    ln_kernel<<<MTOT, 256, 0, stream>>>(x, ln_w, ln_b, h16, hacc);
    scan_kernel<<<1, 256, 0, stream>>>(hacc, h16);
    {
        int waves = (MTOT / 64) * (QKVP_ / 32);  // 14336 single-wave blocks
        gemm_qkvp_kernel<<<waves, 32, 0, stream>>>(h16, wi16, qf32, kf32,
                                                   vT, g2in);
    }
    {
        int n = B_ * L_ * H_ * 32;  // 2^21
        rope_kernel<<<n / 256, 256, 0, stream>>>(qf32, kf32, q16, k16);
    }
    attn_kernel<<<B_ * H_ * (L_ / 16), 32, 0, stream>>>(q16, k16, vT, g2in);
    {
        int waves = (MTOT / 64) * (HID_ / 32);  // 2048 single-wave blocks
        gemm_out_kernel<<<waves, 32, 0, stream>>>(g2in, wo16, b_out, out);
    }
}
